// TransformerDecoderLayerWithAttention_PyTorch_69896297775148
// MI455X (gfx1250) — matrix operand order repaired and checked by reading, not compiled, not run on hardware
//
#include <hip/hip_runtime.h>

// ---------------- types ----------------
typedef __bf16 bf16_t;
typedef bf16_t bf16x16 __attribute__((ext_vector_type(16)));
typedef bf16_t bf16x8  __attribute__((ext_vector_type(8)));
typedef float  f32x8   __attribute__((ext_vector_type(8)));
typedef float  f32x4   __attribute__((ext_vector_type(4)));

union AFrag { bf16x16 v; bf16x8 h[2]; };
union CFrag { f32x8 v; float f[8]; };

// ---------------- problem dims ----------------
static constexpr int Bn = 4, Tn = 1024, TMn = 1024, Dn = 1024, Hn = 16, FFn = 4096, HSn = 64;

// ---------------- tile config ----------------
#define BK  32
#define LAP 40   // lA pitch (bf16 elems): 80B rows (16B multiple)

// epilogue feature bits
#define E_BIAS 1
#define E_RES  2
#define E_RELU 4

// ===========================================================================
// WMMA GEMM:  C[z] = alpha * A[z] @ B[z] (+ bias) (+ R) (ReLU?)
// A: [M,K] row-major (lda). B: [K,N] row-major (ldb). 256 threads = 8 waves
// arranged MWAVES x NWAVES; each wave computes a (WM16*16) x 64 strip of C:
// WM16 A fragments x 4 B fragments -> WM16*4 v_wmma_f32_16x16x32_bf16 per
// K-step (every fragment reused). LDS double-buffered; next tile's global
// loads are issued before current tile's WMMAs.
//   wide   <4,2,2>: block tile 128x128 (M%128==0, N%128==0), 8 wmma/wave/step
//   narrow <8,1,1>: block tile 128x64  (M%128==0, N%64==0),  4 wmma/wave/step
// ===========================================================================
template<int MWAVES, int NWAVES, int WM16, int EPI>
__global__ __launch_bounds__(256)
void gemm_wmma_kernel(const float* __restrict__ A, long long sAz, int lda,
                      const float* __restrict__ Bm, long long sBz, int ldb,
                      const float* __restrict__ bias,
                      const float* __restrict__ R, int ldr,
                      float* __restrict__ C, long long sCz, int ldc,
                      int K, float alpha)
{
    constexpr bool HASBIAS = (EPI & E_BIAS) != 0;
    constexpr bool HASRES  = (EPI & E_RES)  != 0;
    constexpr bool HASRELU = (EPI & E_RELU) != 0;
    constexpr int BMt = MWAVES * WM16 * 16;       // 128
    constexpr int BNt = NWAVES * 64;              // 128 or 64
    constexpr int LBP = BNt + 16;                 // 144 (288B) or 80 (160B)
    constexpr int ACH = (BMt * BK) / (256 * 8);   // 2
    constexpr int BCH = (BK * BNt) / (256 * 8);   // 2 or 1

    __shared__ __align__(16) bf16_t lA[2][BMt * LAP];
    __shared__ __align__(16) bf16_t lB[2][BK * LBP];

    const int t  = threadIdx.x;
    const int M0 = blockIdx.y * BMt;
    const int N0 = blockIdx.x * BNt;
    const int z  = blockIdx.z;
    A  += (long long)z * sAz;
    Bm += (long long)z * sBz;
    C  += (long long)z * sCz;

    const int wave  = t >> 5;
    const int lane  = t & 31;
    const int row   = lane & 15;                      // fragment sub-index
    const int hi    = lane >> 4;                      // half-wave select
    const int Moff  = (wave % MWAVES) * (WM16 * 16);  // wave's M offset
    const int Nbase = (wave / MWAVES) * 64;           // wave's N offset

    CFrag acc[WM16][4];
#pragma unroll
    for (int u = 0; u < WM16; ++u)
#pragma unroll
        for (int s = 0; s < 4; ++s)
#pragma unroll
            for (int i = 0; i < 8; ++i) acc[u][s].f[i] = 0.0f;

    // register-staged tiles (fp32)
    f32x4 ar[ACH][2], br[BCH][2];

    auto loadG = [&](int k0) {
#pragma unroll
        for (int ci = 0; ci < ACH; ++ci) {
            const int idx = (ci * 256 + t) * 8, r = idx >> 5, c = idx & 31;
            const float* src = A + (long long)(M0 + r) * lda + (k0 + c);
            ar[ci][0] = *(const f32x4*)src;
            ar[ci][1] = *(const f32x4*)(src + 4);
        }
#pragma unroll
        for (int ci = 0; ci < BCH; ++ci) {
            const int idx = (ci * 256 + t) * 8, r = idx / BNt, c = idx & (BNt - 1);
            const float* src = Bm + (long long)(k0 + r) * ldb + (N0 + c);
            br[ci][0] = *(const f32x4*)src;
            br[ci][1] = *(const f32x4*)(src + 4);
        }
    };
    auto storeL = [&](int buf) {
#pragma unroll
        for (int ci = 0; ci < ACH; ++ci) {
            const int idx = (ci * 256 + t) * 8, r = idx >> 5, c = idx & 31;
            bf16x8 o;
#pragma unroll
            for (int j = 0; j < 4; ++j) { o[j] = (bf16_t)ar[ci][0][j]; o[4 + j] = (bf16_t)ar[ci][1][j]; }
            *(bf16x8*)&lA[buf][r * LAP + c] = o;
        }
#pragma unroll
        for (int ci = 0; ci < BCH; ++ci) {
            const int idx = (ci * 256 + t) * 8, r = idx / BNt, c = idx & (BNt - 1);
            bf16x8 o;
#pragma unroll
            for (int j = 0; j < 4; ++j) { o[j] = (bf16_t)br[ci][0][j]; o[4 + j] = (bf16_t)br[ci][1][j]; }
            *(bf16x8*)&lB[buf][r * LBP + c] = o;
        }
    };

    loadG(0);
    storeL(0);
    int cur = 0;

    for (int k0 = 0; k0 < K; k0 += BK) {
        __syncthreads();
        const bool more = (k0 + BK) < K;
        if (more) loadG(k0 + BK);       // overlap next tile's VMEM with WMMAs

        // ---- WM16 A fragments + 4 B fragments -> WM16*4 WMMAs ----
        AFrag a[WM16];
#pragma unroll
        for (int u = 0; u < WM16; ++u) {
            const bf16_t* ap = &lA[cur][(Moff + u * 16 + row) * LAP];
            a[u].h[0] = *(const bf16x8*)(ap + 8 * hi);        // K = 8*hi .. +7
            a[u].h[1] = *(const bf16x8*)(ap + 16 + 8 * hi);   // K = 16+8*hi .. +7
        }
        AFrag bfr[4];
#pragma unroll
        for (int s = 0; s < 4; ++s) {
            const bf16_t* bp = &lB[cur][Nbase + s * 16 + row];  // lane%16 = N col, elems over K
#pragma unroll
            for (int j = 0; j < 8; ++j) {
                bfr[s].h[0][j] = bp[(8 * hi + j) * LBP];
                bfr[s].h[1][j] = bp[(16 + 8 * hi + j) * LBP];
            }
        }
#pragma unroll
        for (int u = 0; u < WM16; ++u)
#pragma unroll
            for (int s = 0; s < 4; ++s)
                acc[u][s].v = __builtin_amdgcn_wmma_f32_16x16x32_bf16(
                    false, a[u].v, false, bfr[s].v, (short)0, acc[u][s].v, false, false);

        if (more) { storeL(cur ^ 1); cur ^= 1; }
    }

    // ---- epilogue: alpha, bias, residual, relu (incremental row pointers) ----
#pragma unroll
    for (int u = 0; u < WM16; ++u) {
        const long long mBase = (long long)(M0 + Moff + u * 16 + 8 * hi);
#pragma unroll
        for (int s = 0; s < 4; ++s) {
            const int n = N0 + Nbase + s * 16 + row;
            const float bv = HASBIAS ? bias[n] : 0.0f;
            float* cp = C + mBase * ldc + n;
            const float* rp = nullptr;
            if (HASRES) rp = R + mBase * ldr + n;
#pragma unroll
            for (int r8 = 0; r8 < 8; ++r8) {
                float val = alpha * acc[u][s].f[r8] + bv;
                if (HASRES) { val += *rp; rp += ldr; }
                if (HASRELU) val = fmaxf(val, 0.0f);
                *cp = val;
                cp += ldc;
            }
        }
    }
}

// ===========================================================================
// 32x32 LDS transpose: out[b, d, n] = in[b, n, d]   (in: [B, TMn, Dn])
// ===========================================================================
__global__ __launch_bounds__(256)
void transpose_kernel(const float* __restrict__ in, float* __restrict__ out)
{
    __shared__ float tile[32][33];
    const int tx = threadIdx.x & 31;
    const int ty = threadIdx.x >> 5;          // 0..7
    const int n0 = blockIdx.x * 32;           // TM index
    const int d0 = blockIdx.y * 32;           // D index
    const int b  = blockIdx.z;
    const float* src = in + ((long long)b * TMn + n0) * Dn + d0;
#pragma unroll
    for (int r = 0; r < 32; r += 8)
        tile[ty + r][tx] = src[(long long)(ty + r) * Dn + tx];
    __syncthreads();
    float* dst = out + ((long long)b * Dn + d0) * TMn + n0;
#pragma unroll
    for (int r = 0; r < 32; r += 8)
        dst[(long long)(ty + r) * TMn + tx] = tile[tx][ty + r];
}

// ===========================================================================
// Row softmax (cols == 1024), optional fused accumulation of head-mean probs
// ===========================================================================
__global__ __launch_bounds__(256)
void softmax_kernel(float* __restrict__ S, float* __restrict__ accum,
                    int cols, float accumScale)
{
    const long long base = (long long)blockIdx.x * cols;
    const int t = threadIdx.x;
    float* p = S + base;
    f32x4 v = *(const f32x4*)(p + t * 4);

    __shared__ float red[8];
    __shared__ float bcast;

    float mx = fmaxf(fmaxf(v[0], v[1]), fmaxf(v[2], v[3]));
    for (int off = 16; off > 0; off >>= 1) mx = fmaxf(mx, __shfl_xor(mx, off, 32));
    if ((t & 31) == 0) red[t >> 5] = mx;
    __syncthreads();
    if (t == 0) { float m = red[0]; for (int i = 1; i < 8; ++i) m = fmaxf(m, red[i]); bcast = m; }
    __syncthreads();
    mx = bcast;

    float s = 0.0f;
#pragma unroll
    for (int j = 0; j < 4; ++j) { v[j] = __expf(v[j] - mx); s += v[j]; }
    for (int off = 16; off > 0; off >>= 1) s += __shfl_xor(s, off, 32);
    __syncthreads();
    if ((t & 31) == 0) red[t >> 5] = s;
    __syncthreads();
    if (t == 0) { float m = 0.0f; for (int i = 0; i < 8; ++i) m += red[i]; bcast = m; }
    __syncthreads();
    const float inv = 1.0f / bcast;

#pragma unroll
    for (int j = 0; j < 4; ++j) v[j] *= inv;
    *(f32x4*)(p + t * 4) = v;

    if (accum) {
        float* ap = accum + base;
        f32x4 av = *(const f32x4*)(ap + t * 4);
#pragma unroll
        for (int j = 0; j < 4; ++j) av[j] += v[j] * accumScale;
        *(f32x4*)(ap + t * 4) = av;
    }
}

// ===========================================================================
// Row LayerNorm (cols == 1024)
// ===========================================================================
__global__ __launch_bounds__(256)
void layernorm_kernel(const float* __restrict__ X, const float* __restrict__ g,
                      const float* __restrict__ b, float* __restrict__ O, int cols)
{
    const long long base = (long long)blockIdx.x * cols;
    const int t = threadIdx.x;
    f32x4 v = *(const f32x4*)(X + base + t * 4);

    float s  = v[0] + v[1] + v[2] + v[3];
    float sq = v[0]*v[0] + v[1]*v[1] + v[2]*v[2] + v[3]*v[3];

    __shared__ float redS[8], redQ[8];
    __shared__ float mBc, iBc;
    for (int off = 16; off > 0; off >>= 1) { s += __shfl_xor(s, off, 32); sq += __shfl_xor(sq, off, 32); }
    if ((t & 31) == 0) { redS[t >> 5] = s; redQ[t >> 5] = sq; }
    __syncthreads();
    if (t == 0) {
        float ts = 0.0f, tq = 0.0f;
        for (int i = 0; i < 8; ++i) { ts += redS[i]; tq += redQ[i]; }
        const float mean = ts / (float)cols;
        const float var  = tq / (float)cols - mean * mean;
        mBc = mean;
        iBc = rsqrtf(var + 1e-5f);
    }
    __syncthreads();
    const float mean = mBc, inv = iBc;

    f32x4 gv = *(const f32x4*)(g + t * 4);
    f32x4 bv = *(const f32x4*)(b + t * 4);
    f32x4 o;
#pragma unroll
    for (int j = 0; j < 4; ++j) o[j] = (v[j] - mean) * inv * gv[j] + bv[j];
    *(f32x4*)(O + base + t * 4) = o;
}

__global__ void zero_kernel(float* __restrict__ p, long long n)
{
    long long i = (long long)blockIdx.x * blockDim.x + threadIdx.x;
    const long long stride = (long long)gridDim.x * blockDim.x;
    for (; i < n; i += stride) p[i] = 0.0f;
}

// ===========================================================================
// Host-side orchestration
// ===========================================================================
extern "C" void kernel_launch(void* const* d_in, const int* in_sizes, int n_in,
                              void* d_out, int out_size, void* d_ws, size_t ws_size,
                              hipStream_t stream)
{
    (void)in_sizes; (void)n_in; (void)out_size; (void)ws_size;

    const float* tgt   = (const float*)d_in[0];
    const float* mem   = (const float*)d_in[1];
    const float* sa_wq = (const float*)d_in[2];  const float* sa_bq = (const float*)d_in[3];
    const float* sa_wk = (const float*)d_in[4];  const float* sa_bk = (const float*)d_in[5];
    const float* sa_wv = (const float*)d_in[6];  const float* sa_bv = (const float*)d_in[7];
    const float* sa_wo = (const float*)d_in[8];  const float* sa_bo = (const float*)d_in[9];
    const float* ca_wq = (const float*)d_in[10]; const float* ca_bq = (const float*)d_in[11];
    const float* ca_wk = (const float*)d_in[12]; const float* ca_bk = (const float*)d_in[13];
    const float* ca_wv = (const float*)d_in[14]; const float* ca_bv = (const float*)d_in[15];
    const float* ca_wo = (const float*)d_in[16]; const float* ca_bo = (const float*)d_in[17];
    const float* w1    = (const float*)d_in[18]; const float* b1    = (const float*)d_in[19];
    const float* w2    = (const float*)d_in[20]; const float* b2    = (const float*)d_in[21];
    const float* ln1g  = (const float*)d_in[22]; const float* ln1b  = (const float*)d_in[23];
    const float* ln2g  = (const float*)d_in[24]; const float* ln2b  = (const float*)d_in[25];
    const float* ln3g  = (const float*)d_in[26]; const float* ln3b  = (const float*)d_in[27];

    const size_t nBTD = (size_t)Bn * Tn * Dn;       // 4M floats
    float* xout    = (float*)d_out;                 // [B,T,D]
    float* attnout = xout + nBTD;                   // [B,T,TM]

    float* ws = (float*)d_ws;
    float* qb = ws + 0 * nBTD;
    float* kb = ws + 1 * nBTD;
    float* vb = ws + 2 * nBTD;
    float* yb = ws + 3 * nBTD;   // concat of per-head attention outputs
    float* sb = ws + 4 * nBTD;   // residual sums (pre-LN)
    float* xb = ws + 5 * nBTD;   // post-LN activations
    float* sc = ws + 6 * nBTD;   // per-head scores [B,T,TM]
    float* fh = ws + 7 * nBTD;   // FFN hidden [B,T,FF] (4*nBTD)
    float* kt = ws + 11 * nBTD;  // transposed K: [B, D, TM]

    const dim3 blk(256);
    const int MT = Bn * Tn;      // 4096

    // wide <4,2,2>: block 128x128; narrow <8,1,1>: block 128x64 (for N=64)
    auto gemm = [&](const float* Ap, long long sAz, int lda,
                    const float* Bp, long long sBz, int ldb,
                    const float* bias,
                    const float* Rp, int ldr,
                    float* Cp, long long sCz, int ldc,
                    int M, int N, int Kd, int batch, float alpha,
                    bool narrow, bool relu) {
        if (narrow) {
            dim3 grid(N / 64, M / 128, batch);
            gemm_wmma_kernel<8, 1, 1, 0><<<grid, blk, 0, stream>>>(
                Ap, sAz, lda, Bp, sBz, ldb, nullptr, nullptr, 0, Cp, sCz, ldc, Kd, alpha);
        } else {
            dim3 grid(N / 128, M / 128, batch);
            if (relu)
                gemm_wmma_kernel<4, 2, 2, E_BIAS | E_RELU><<<grid, blk, 0, stream>>>(
                    Ap, sAz, lda, Bp, sBz, ldb, bias, nullptr, 0, Cp, sCz, ldc, Kd, alpha);
            else if (Rp)
                gemm_wmma_kernel<4, 2, 2, E_BIAS | E_RES><<<grid, blk, 0, stream>>>(
                    Ap, sAz, lda, Bp, sBz, ldb, bias, Rp, ldr, Cp, sCz, ldc, Kd, alpha);
            else if (bias)
                gemm_wmma_kernel<4, 2, 2, E_BIAS><<<grid, blk, 0, stream>>>(
                    Ap, sAz, lda, Bp, sBz, ldb, bias, nullptr, 0, Cp, sCz, ldc, Kd, alpha);
            else
                gemm_wmma_kernel<4, 2, 2, 0><<<grid, blk, 0, stream>>>(
                    Ap, sAz, lda, Bp, sBz, ldb, nullptr, nullptr, 0, Cp, sCz, ldc, Kd, alpha);
        }
    };

    const float invSqrtHS = 0.125f;           // 1/sqrt(64)
    const long long sTD  = (long long)Tn * Dn;
    const long long sTTM = (long long)Tn * TMn;
    const long long sDTM = (long long)Dn * TMn;
    const dim3 trgrid(TMn / 32, Dn / 32, Bn);

    // ================= self-attention =================
    gemm(tgt, 0, Dn, sa_wq, 0, Dn, sa_bq, nullptr, 0, qb, 0, Dn, MT, Dn, Dn, 1, 1.0f, false, false);
    gemm(tgt, 0, Dn, sa_wk, 0, Dn, sa_bk, nullptr, 0, kb, 0, Dn, MT, Dn, Dn, 1, 1.0f, false, false);
    gemm(tgt, 0, Dn, sa_wv, 0, Dn, sa_bv, nullptr, 0, vb, 0, Dn, MT, Dn, Dn, 1, 1.0f, false, false);
    transpose_kernel<<<trgrid, blk, 0, stream>>>(kb, kt);   // kt[b,d,n] = kb[b,n,d]
    for (int h = 0; h < Hn; ++h) {
        // S = (Qh Kh^T) / sqrt(HS): B tile is kt rows h*HS..h*HS+63, row-major over n
        gemm(qb + h * HSn, sTD, Dn, kt + (long long)h * HSn * TMn, sDTM, TMn,
             nullptr, nullptr, 0, sc, sTTM, TMn, Tn, TMn, HSn, Bn, invSqrtHS, false, false);
        softmax_kernel<<<Bn * Tn, blk, 0, stream>>>(sc, nullptr, TMn, 0.0f);
        // Yh = P Vh  (N = 64 -> narrow config)
        gemm(sc, sTTM, TMn, vb + h * HSn, sTD, Dn, nullptr, nullptr, 0,
             yb + h * HSn, sTD, Dn, Tn, HSn, TMn, Bn, 1.0f, true, false);
    }
    gemm(yb, 0, Dn, sa_wo, 0, Dn, sa_bo, tgt, Dn, sb, 0, Dn, MT, Dn, Dn, 1, 1.0f, false, false);
    layernorm_kernel<<<MT, blk, 0, stream>>>(sb, ln1g, ln1b, xb, Dn);

    // ================= cross-attention =================
    gemm(xb,  0, Dn, ca_wq, 0, Dn, ca_bq, nullptr, 0, qb, 0, Dn, MT, Dn, Dn, 1, 1.0f, false, false);
    gemm(mem, 0, Dn, ca_wk, 0, Dn, ca_bk, nullptr, 0, kb, 0, Dn, Bn * TMn, Dn, Dn, 1, 1.0f, false, false);
    gemm(mem, 0, Dn, ca_wv, 0, Dn, ca_bv, nullptr, 0, vb, 0, Dn, Bn * TMn, Dn, Dn, 1, 1.0f, false, false);
    transpose_kernel<<<trgrid, blk, 0, stream>>>(kb, kt);
    zero_kernel<<<2048, blk, 0, stream>>>(attnout, (long long)Bn * Tn * TMn);
    for (int h = 0; h < Hn; ++h) {
        gemm(qb + h * HSn, sTD, Dn, kt + (long long)h * HSn * TMn, sDTM, TMn,
             nullptr, nullptr, 0, sc, sTTM, TMn, Tn, TMn, HSn, Bn, invSqrtHS, false, false);
        softmax_kernel<<<Bn * Tn, blk, 0, stream>>>(sc, attnout, TMn, 1.0f / (float)Hn);
        gemm(sc, sTTM, TMn, vb + h * HSn, sTD, Dn, nullptr, nullptr, 0,
             yb + h * HSn, sTD, Dn, Tn, HSn, TMn, Bn, 1.0f, true, false);
    }
    gemm(yb, 0, Dn, ca_wo, 0, Dn, ca_bo, xb, Dn, sb, 0, Dn, MT, Dn, Dn, 1, 1.0f, false, false);
    layernorm_kernel<<<MT, blk, 0, stream>>>(sb, ln2g, ln2b, xb, Dn);

    // ================= FFN =================
    gemm(xb, 0, Dn, w1, 0, FFn, b1, nullptr, 0, fh, 0, FFn, MT, FFn, Dn, 1, 1.0f, false, true);
    gemm(fh, 0, FFn, w2, 0, Dn, b2, xb, Dn, sb, 0, Dn, MT, Dn, FFn, 1, 1.0f, false, false);
    layernorm_kernel<<<MT, blk, 0, stream>>>(sb, ln3g, ln3b, xout, Dn);
}
